// Mamba_58128087384453
// MI455X (gfx1250) — matrix operand order repaired and checked by reading, not compiled, not run on hardware
//
#include <hip/hip_runtime.h>
#include <math.h>

// ---------------------------------------------------------------------------
// Mamba forward for MI455X (gfx1250), wave32.
//  - GEMMs (~343 GFLOP total) on v_wmma_f32_16x16x32_bf16 (f32 accumulate)
//  - Operand tiles staged global->LDS by the Tensor Data Mover
//    (tensor_load_to_lds + s_wait_tensorcnt), double-buffered, with D#
//    LDS padding for conflict-free ds_load_b128 fragment reads.
// Shapes: BATCH=2, SEQ=2048, D_MODEL=2048, D_INNER=4096, D_STATE=16, D_CONV=4
// ---------------------------------------------------------------------------

#define BATCH   2
#define SEQ     2048
#define DMODEL  2048
#define DINNER  4096
#define DSTATE  16
#define DCONV   4
#define NTOK    (BATCH * SEQ)        // 4096

typedef __attribute__((ext_vector_type(16))) __bf16        v16bf;
typedef __attribute__((ext_vector_type(8)))  float         v8f;
typedef __attribute__((ext_vector_type(4)))  unsigned int  u32x4;
typedef __attribute__((ext_vector_type(8)))  int           i32x8;
typedef __attribute__((ext_vector_type(4)))  int           i32x4;

// ---- f32 <-> bf16 helpers (round-to-nearest-even on the way down) ---------
__device__ __forceinline__ unsigned short f2bf(float f) {
  unsigned u = __float_as_uint(f);
  u += 0x7FFFu + ((u >> 16) & 1u);
  return (unsigned short)(u >> 16);
}
__device__ __forceinline__ float bf2f(unsigned short h) {
  return __uint_as_float(((unsigned)h) << 16);
}

// ---------------------------------------------------------------------------
// K0: elementwise f32 -> bf16 conversion
// ---------------------------------------------------------------------------
__global__ void cvt_f32_bf16(const float* __restrict__ in,
                             unsigned short* __restrict__ out, int n) {
  int i = blockIdx.x * blockDim.x + threadIdx.x;
  if (i < n) out[i] = f2bf(in[i]);
}

// ---------------------------------------------------------------------------
// TDM: issue one tensor_load_to_lds moving a 128-row x 32-col bf16 tile
// (row pitch `ld` elements in global) into LDS at byte offset lds_off.
// LDS padding: pad_interval=3 (pad after every 16 DWORDs = 64B = one row),
// pad_amount=3 (4 DWORDs = 16B)  ->  LDS row pitch 80B, so ds_load_b128
// fragment reads (16 lanes, rows r, banks 20*r mod 64) are conflict-free.
// Tile LDS footprint: 128 * 80B = 10240B.
// D# bitfields per CDNA5 ISA ch.8 (group0: count/lds/global/type,
// group1: data_size/pads/dims/strides).
// ---------------------------------------------------------------------------
#define TILE_LDS_BYTES 10240

__device__ __forceinline__ void tdm_load_tile(const unsigned short* gsrc,
                                              unsigned lds_off,
                                              unsigned rows, unsigned ld) {
  unsigned long long ga = (unsigned long long)gsrc;
  u32x4 g0;
  g0[0] = 1u;                                        // count=1, user descriptor
  g0[1] = lds_off;                                   // lds_addr (bytes)
  g0[2] = (unsigned)(ga & 0xFFFFFFFFu);              // global_addr[31:0]
  g0[3] = (unsigned)((ga >> 32) & 0x01FFFFFFu)       // global_addr[56:32]
        | (2u << 30);                                // type = 2 ("image")

  i32x8 g1;
  g1[0] = (int)((1u << 16)                           // data_size = 1 (2 bytes)
              | (1u << 20)                           // pad_enable
              | (3u << 22)                           // pad_interval: 16 DWORDs
              | (3u << 25));                         // pad_amount: 4 DWORDs
  g1[1] = (int)((ld & 0xFFFFu) << 16);               // tensor_dim0[15:0]
  g1[2] = (int)(((ld >> 16) & 0xFFFFu)               // tensor_dim0[31:16]
              | ((rows & 0xFFFFu) << 16));           // tensor_dim1[15:0]
  g1[3] = (int)(((rows >> 16) & 0xFFFFu)             // tensor_dim1[31:16]
              | (32u << 16));                        // tile_dim0 = 32 elems
  g1[4] = (int)128u;                                 // tile_dim1 = 128 rows
  g1[5] = (int)ld;                                   // tensor_dim0_stride[31:0]
  g1[6] = 0;                                         // stride0[47:32], stride1 lo
  g1[7] = 0;                                         // stride1 hi (unused, 2D)

  i32x4 z4 = {0, 0, 0, 0};
#if defined(__clang_major__) && (__clang_major__ >= 23)
  i32x8 z8 = {0, 0, 0, 0, 0, 0, 0, 0};
  __builtin_amdgcn_tensor_load_to_lds(g0, g1, z4, z4, z8, 0);
#else
  __builtin_amdgcn_tensor_load_to_lds(g0, g1, z4, z4, 0);
#endif
}

// ---------------------------------------------------------------------------
// WMMA fragment loads from an LDS tile (row pitch 40 shorts = 80B).
// A matrix 16x32 bf16 (ISA 7.12.2): lane L -> row L&15, two 16B chunks at
//   K = (L>>4)*8 and +16.
// B matrix 32x16 bf16: lane L -> column L&15, two 16B chunks at
//   K = (L>>4)*8 and +16 (same K order as A).
// Our B source rows are rows of W (N x K), i.e. columns of B.
// ---------------------------------------------------------------------------
union FragU { v16bf v; u32x4 q[2]; };

__device__ __forceinline__ v16bf lds_fragA(const unsigned short* t, int row0, int lane) {
  const int idx = (row0 + (lane & 15)) * 40 + (lane >> 4) * 8;
  FragU f;
  f.q[0] = *(const u32x4*)(t + idx);
  f.q[1] = *(const u32x4*)(t + idx + 16);
  return f.v;
}
__device__ __forceinline__ v16bf lds_fragB(const unsigned short* t, int row0, int lane) {
  const int idx = (row0 + (lane & 15)) * 40 + (lane >> 4) * 8;
  FragU f;
  f.q[0] = *(const u32x4*)(t + idx);
  f.q[1] = *(const u32x4*)(t + idx + 16);
  return f.v;
}

// ---------------------------------------------------------------------------
// Tiled WMMA GEMM:  C[M,N] = A[M,K] * B[N,K]^T   (A, B bf16; accumulate f32)
//   Workgroup: 256 threads = 8 waves (4 x 2), WG tile 128(M) x 128(N)
//   Wave tile: 32(M) x 64(N) = 2x4 v_wmma_f32_16x16x32_bf16 per 32-K step.
//   A/B tiles TDM-staged into LDS, double buffered on TENSORcnt.
// MODE 0: store bf16(acc)                           (in_proj -> xz)
// MODE 1: store bf16(exp(softplus(acc+bias[n])))    == 1 + exp(acc+bias)
// MODE 2: store f32(acc)                            (out_proj -> d_out)
// ---------------------------------------------------------------------------
template <int MODE>
__global__ __launch_bounds__(256)
void gemm_bf16_wmma(const unsigned short* __restrict__ A,
                    const unsigned short* __restrict__ B,
                    void* __restrict__ Cout,
                    const float* __restrict__ bias,
                    int M, int N, int K, int lda, int ldb, int ldc) {
  // LDS: [A buf0 | B buf0 | A buf1 | B buf1], 10240B each (assumed at offset 0
  // of the workgroup LDS allocation; it is the only __shared__ object).
  __shared__ __align__(16) unsigned short smem[4 * (TILE_LDS_BYTES / 2)];

  const int lane = threadIdx.x & 31;
  const int wave = threadIdx.x >> 5;
  const int wm = wave >> 1;                // 0..3
  const int wn = wave & 1;                 // 0..1
  const int mblk = blockIdx.y * 128;
  const int nblk = blockIdx.x * 128;
  const int m0 = mblk + wm * 32;
  const int n0 = nblk + wn * 64;

  v8f acc[2][4];
#pragma unroll
  for (int i = 0; i < 2; ++i)
#pragma unroll
    for (int j = 0; j < 4; ++j) acc[i][j] = (v8f)0.0f;

  const unsigned short* Atile = A + (size_t)mblk * lda;
  const unsigned short* Btile = B + (size_t)nblk * ldb;
  const int nk = K / 32;

  // Prologue: stage 0 into buffer 0 (wave 0 drives the TDM; EXEC is ignored
  // by tensor ops and the descriptor operands are wave-uniform SGPRs).
  if (wave == 0) {
    tdm_load_tile(Atile, 0u,                  128u, (unsigned)lda);
    tdm_load_tile(Btile, TILE_LDS_BYTES,      128u, (unsigned)ldb);
  }

  for (int i = 0; i < nk; ++i) {
    const int cur = i & 1;
    if (wave == 0) {
      if (i + 1 < nk) {
        const unsigned nb = (unsigned)((i + 1) & 1) * (2 * TILE_LDS_BYTES);
        tdm_load_tile(Atile + (size_t)(i + 1) * 32, nb,                  128u, (unsigned)lda);
        tdm_load_tile(Btile + (size_t)(i + 1) * 32, nb + TILE_LDS_BYTES, 128u, (unsigned)ldb);
        // Tensor ops complete in order: <=2 outstanding => stage i finished.
        __builtin_amdgcn_s_wait_tensorcnt(2);
      } else {
        __builtin_amdgcn_s_wait_tensorcnt(0);
      }
    }
    __syncthreads();  // stage i visible to all waves

    const unsigned short* At = smem + cur * TILE_LDS_BYTES;             // /2*2
    const unsigned short* Bt = smem + (TILE_LDS_BYTES / 2) + cur * TILE_LDS_BYTES;

    v16bf a0 = lds_fragA(At, wm * 32,      lane);
    v16bf a1 = lds_fragA(At, wm * 32 + 16, lane);
    v16bf b0 = lds_fragB(Bt, wn * 64,      lane);
    v16bf b1 = lds_fragB(Bt, wn * 64 + 16, lane);
    v16bf b2 = lds_fragB(Bt, wn * 64 + 32, lane);
    v16bf b3 = lds_fragB(Bt, wn * 64 + 48, lane);

    acc[0][0] = __builtin_amdgcn_wmma_f32_16x16x32_bf16(false, a0, false, b0, (short)0, acc[0][0], false, false);
    acc[0][1] = __builtin_amdgcn_wmma_f32_16x16x32_bf16(false, a0, false, b1, (short)0, acc[0][1], false, false);
    acc[0][2] = __builtin_amdgcn_wmma_f32_16x16x32_bf16(false, a0, false, b2, (short)0, acc[0][2], false, false);
    acc[0][3] = __builtin_amdgcn_wmma_f32_16x16x32_bf16(false, a0, false, b3, (short)0, acc[0][3], false, false);
    acc[1][0] = __builtin_amdgcn_wmma_f32_16x16x32_bf16(false, a1, false, b0, (short)0, acc[1][0], false, false);
    acc[1][1] = __builtin_amdgcn_wmma_f32_16x16x32_bf16(false, a1, false, b1, (short)0, acc[1][1], false, false);
    acc[1][2] = __builtin_amdgcn_wmma_f32_16x16x32_bf16(false, a1, false, b2, (short)0, acc[1][2], false, false);
    acc[1][3] = __builtin_amdgcn_wmma_f32_16x16x32_bf16(false, a1, false, b3, (short)0, acc[1][3], false, false);

    __syncthreads();  // everyone done reading buf[cur] before it is re-filled
  }

  // D layout (32-bit C/D 16x16): VGPR v -> M = v + 8*(lane>>4), N = lane&15
  const int mrow = (lane >> 4) * 8;
  const int ncol = lane & 15;
#pragma unroll
  for (int i = 0; i < 2; ++i) {
#pragma unroll
    for (int j = 0; j < 4; ++j) {
#pragma unroll
      for (int v = 0; v < 8; ++v) {
        const int m = m0 + 16 * i + mrow + v;
        const int n = n0 + 16 * j + ncol;
        float val = acc[i][j][v];
        if (MODE == 1) {
          // exp(softplus(x)) == 1 + exp(x) exactly; clamp to avoid inf
          val = 1.0f + expf(fminf(val + bias[n], 80.0f));
        }
        if (MODE == 2) {
          ((float*)Cout)[(size_t)m * ldc + n] = val;
        } else {
          ((unsigned short*)Cout)[(size_t)m * ldc + n] = f2bf(val);
        }
      }
    }
  }
}

// ---------------------------------------------------------------------------
// K2: causal depthwise conv (kernel 4, left pad 3, per-batch boundaries)
//   reads xc = columns [0, DINNER) of xz (row stride 2*DINNER), writes xcv
// ---------------------------------------------------------------------------
__global__ void conv_kernel(const unsigned short* __restrict__ xz,
                            const float* __restrict__ cw,
                            const float* __restrict__ cb,
                            unsigned short* __restrict__ xcv) {
  const int idx = blockIdx.x * blockDim.x + threadIdx.x;
  if (idx >= NTOK * DINNER) return;
  const int c = idx & (DINNER - 1);
  const int t = idx >> 12;               // DINNER == 4096 == 2^12
  const int l = t & (SEQ - 1);
  float acc = cb[c];
#pragma unroll
  for (int j = 0; j < DCONV; ++j) {
    const int ll = l - (DCONV - 1) + j;
    if (ll >= 0) {
      const int tt = t + (ll - l);
      acc += cw[c * DCONV + j] * bf2f(xz[(size_t)tt * (2 * DINNER) + c]);
    }
  }
  xcv[(size_t)t * DINNER + c] = f2bf(acc);
}

// ---------------------------------------------------------------------------
// K3: skinny reductions over d_inner (K=4096, N=16):
//   dA[t,s] = sum_c dexp[t,c] * (-exp(A_log[c,s]))
//   xB[t,s] = sum_c xcv[t,c]  * B_ssm[c,s]
// One block per token; LDS float-atomic reduction (ds_add_f32).
// ---------------------------------------------------------------------------
__global__ __launch_bounds__(256)
void dA_xB_kernel(const unsigned short* __restrict__ dexp,
                  const unsigned short* __restrict__ xcv,
                  const float* __restrict__ A_log,
                  const float* __restrict__ Bs,
                  float* __restrict__ dA, float* __restrict__ xB) {
  const int t = blockIdx.x;
  const int tid = threadIdx.x;
  __shared__ float sA[DSTATE];
  __shared__ float sB[DSTATE];
  if (tid < DSTATE) { sA[tid] = 0.0f; sB[tid] = 0.0f; }
  __syncthreads();

  float accA[DSTATE], accB[DSTATE];
#pragma unroll
  for (int s = 0; s < DSTATE; ++s) { accA[s] = 0.0f; accB[s] = 0.0f; }

  for (int c = tid; c < DINNER; c += 256) {
    const float de = bf2f(dexp[(size_t)t * DINNER + c]);
    const float xv = bf2f(xcv[(size_t)t * DINNER + c]);
#pragma unroll
    for (int s = 0; s < DSTATE; ++s) {
      accA[s] += de * (-expf(A_log[c * DSTATE + s]));
      accB[s] += xv * Bs[c * DSTATE + s];
    }
  }
#pragma unroll
  for (int s = 0; s < DSTATE; ++s) {
    atomicAdd(&sA[s], accA[s]);
    atomicAdd(&sB[s], accB[s]);
  }
  __syncthreads();
  if (tid < DSTATE) {
    dA[(size_t)t * DSTATE + tid] = sA[tid];
    xB[(size_t)t * DSTATE + tid] = sB[tid];
  }
}

// ---------------------------------------------------------------------------
// K4: sequential scan.  Only B*DSTATE = 32 independent chains of length SEQ
// (state is (B, d_state) in this model) -> one wave32, one chain per lane.
// ---------------------------------------------------------------------------
__global__ void scan_kernel(const float* __restrict__ dA,
                            const float* __restrict__ xB,
                            float* __restrict__ states) {
  const int i = threadIdx.x;             // 0..31
  const int b = i >> 4;
  const int s = i & 15;
  float st = 0.0f;
  const int base = b * SEQ;
  for (int l = 0; l < SEQ; ++l) {
    const int t = base + l;
    st = fmaf(dA[(size_t)t * DSTATE + s], st, xB[(size_t)t * DSTATE + s]);
    states[(size_t)t * DSTATE + s] = st;
  }
}

// ---------------------------------------------------------------------------
// K5: y = states @ C^T + D*xcv;  y *= silu(z);  store bf16 for out_proj
// ---------------------------------------------------------------------------
__global__ void y_kernel(const float* __restrict__ states,
                         const float* __restrict__ C,
                         const float* __restrict__ D,
                         const unsigned short* __restrict__ xcv,
                         const unsigned short* __restrict__ xz,
                         unsigned short* __restrict__ ybf) {
  const int idx = blockIdx.x * blockDim.x + threadIdx.x;
  if (idx >= NTOK * DINNER) return;
  const int c = idx & (DINNER - 1);
  const int t = idx >> 12;
  float acc = D[c] * bf2f(xcv[(size_t)t * DINNER + c]);
#pragma unroll
  for (int s = 0; s < DSTATE; ++s)
    acc += states[(size_t)t * DSTATE + s] * C[c * DSTATE + s];
  const float z = bf2f(xz[(size_t)t * (2 * DINNER) + DINNER + c]);
  acc *= z / (1.0f + expf(-z));          // z * sigmoid(z)
  ybf[(size_t)t * DINNER + c] = f2bf(acc);
}

// ---------------------------------------------------------------------------
// Host-side orchestration (graph-capture safe: only kernel launches on stream)
// ---------------------------------------------------------------------------
extern "C" void kernel_launch(void* const* d_in, const int* in_sizes, int n_in,
                              void* d_out, int out_size, void* d_ws, size_t ws_size,
                              hipStream_t stream) {
  const float* x      = (const float*)d_in[0];
  const float* W_in   = (const float*)d_in[1];
  const float* conv_w = (const float*)d_in[2];
  const float* conv_b = (const float*)d_in[3];
  const float* W_dt   = (const float*)d_in[4];
  const float* b_dt   = (const float*)d_in[5];
  const float* A_log  = (const float*)d_in[6];
  const float* B_ssm  = (const float*)d_in[7];
  const float* C_ssm  = (const float*)d_in[8];
  const float* D_ssm  = (const float*)d_in[9];
  const float* W_out  = (const float*)d_in[10];

  // Workspace carve-out (~270 MB total), 256B aligned regions.
  char* p = (char*)d_ws;
  auto carve = [&](size_t bytes) -> void* {
    void* r = (void*)p;
    p += (bytes + 255) & ~(size_t)255;
    return r;
  };
  unsigned short* x_bf    = (unsigned short*)carve((size_t)NTOK * DMODEL * 2);
  unsigned short* Win_bf  = (unsigned short*)carve((size_t)2 * DINNER * DMODEL * 2);
  unsigned short* Wdt_bf  = (unsigned short*)carve((size_t)DINNER * DINNER * 2);
  unsigned short* Wout_bf = (unsigned short*)carve((size_t)DMODEL * DINNER * 2);
  unsigned short* xz_bf   = (unsigned short*)carve((size_t)NTOK * 2 * DINNER * 2);
  unsigned short* xcv_bf  = (unsigned short*)carve((size_t)NTOK * DINNER * 2);
  unsigned short* dexp_bf = (unsigned short*)carve((size_t)NTOK * DINNER * 2);
  unsigned short* y_bf    = (unsigned short*)carve((size_t)NTOK * DINNER * 2);
  float* dA     = (float*)carve((size_t)NTOK * DSTATE * 4);
  float* xB     = (float*)carve((size_t)NTOK * DSTATE * 4);
  float* states = (float*)carve((size_t)NTOK * DSTATE * 4);

  auto cvt = [&](const float* in, unsigned short* out, size_t n) {
    cvt_f32_bf16<<<(unsigned)((n + 255) / 256), 256, 0, stream>>>(in, out, (int)n);
  };
  cvt(x,     x_bf,    (size_t)NTOK * DMODEL);
  cvt(W_in,  Win_bf,  (size_t)2 * DINNER * DMODEL);
  cvt(W_dt,  Wdt_bf,  (size_t)DINNER * DINNER);
  cvt(W_out, Wout_bf, (size_t)DMODEL * DINNER);

  // GEMM1: xz = x @ W_in^T   (4096 x 8192, K=2048)
  gemm_bf16_wmma<0><<<dim3((2 * DINNER) / 128, NTOK / 128), 256, 0, stream>>>(
      x_bf, Win_bf, xz_bf, nullptr,
      NTOK, 2 * DINNER, DMODEL, DMODEL, DMODEL, 2 * DINNER);

  // K2: causal depthwise conv + bias -> xcv
  conv_kernel<<<(NTOK * DINNER) / 256, 256, 0, stream>>>(xz_bf, conv_w, conv_b, xcv_bf);

  // GEMM2: dexp = exp(softplus(xcv @ W_dt^T + b_dt))   (4096 x 4096, K=4096)
  gemm_bf16_wmma<1><<<dim3(DINNER / 128, NTOK / 128), 256, 0, stream>>>(
      xcv_bf, Wdt_bf, dexp_bf, b_dt,
      NTOK, DINNER, DINNER, DINNER, DINNER, DINNER);

  // K3: dA = dexp @ (-exp(A_log)),  xB = xcv @ B_ssm
  dA_xB_kernel<<<NTOK, 256, 0, stream>>>(dexp_bf, xcv_bf, A_log, B_ssm, dA, xB);

  // K4: sequential state scan (32 chains)
  scan_kernel<<<1, 32, 0, stream>>>(dA, xB, states);

  // K5: y = (states @ C^T + D*xcv) * silu(z)
  y_kernel<<<(NTOK * DINNER) / 256, 256, 0, stream>>>(states, C_ssm, D_ssm,
                                                      xcv_bf, xz_bf, y_bf);

  // GEMM3: out = y @ W_out^T  (4096 x 2048, K=4096) -> f32 d_out
  gemm_bf16_wmma<2><<<dim3(DMODEL / 128, NTOK / 128), 256, 0, stream>>>(
      y_bf, Wout_bf, (float*)d_out, nullptr,
      NTOK, DMODEL, DINNER, DINNER, DINNER, DMODEL);
}
